// ExplainableIndexer_45088566674078
// MI455X (gfx1250) — matrix operand order repaired and checked by reading, not compiled, not run on hardware
//
#include <hip/hip_runtime.h>

typedef __attribute__((ext_vector_type(16))) __bf16 v16bf;
typedef __attribute__((ext_vector_type(8)))  __bf16 v8bf;
typedef __attribute__((ext_vector_type(8)))  float  v8f;
typedef __attribute__((vector_size(16)))     int    v4i;

#define AS1 __attribute__((address_space(1)))
#define AS3 __attribute__((address_space(3)))

#define B_    4
#define S_    4096
#define DIM_  2048
#define H_    16
#define HD_   128
#define RD_   64
#define QLR_  512
#define ROWS  (B_ * S_)          // 16384
#define KSTEP 128                // K staged per barrier (4 WMMA-K slabs of 32)
#define FWHT_SCALE 0.08838834764831845f   // 128^-0.5

#if __has_builtin(__builtin_amdgcn_global_store_async_from_lds_b128)
#define HAVE_ASYNC_ST 1
#else
#define HAVE_ASYNC_ST 0
#endif

__device__ __forceinline__ void async_drain() {
#if __has_builtin(__builtin_amdgcn_s_wait_asynccnt)
    __builtin_amdgcn_s_wait_asynccnt(0);
#else
    asm volatile("s_wait_asynccnt 0x0" ::: "memory");
#endif
}

// ---------------------------------------------------------------------------
// Weight pre-pass: f32 row-major [K][N] -> bf16 blocked fragment layout
//   wt[((k>>5)*N + n)*32 + (k&31)]
// A lane's 16-value B fragment is contiguous 32B; a wave's fragments span 1KB.
// ---------------------------------------------------------------------------
__global__ void convert_w_kernel(const float* __restrict__ w, __bf16* __restrict__ wt,
                                 int K, int N) {
    size_t i = (size_t)blockIdx.x * blockDim.x + threadIdx.x;
    if (i >= (size_t)K * N) return;
    int k = (int)(i / N), n = (int)(i % N);
    wt[((size_t)(k >> 5) * N + n) * 32 + (k & 31)] = (__bf16)w[i];
}

// A staging: 8-aligned k-group is contiguous in per-lane fragment layout.
__device__ __forceinline__ int a_lds_group_idx(int m, int k /*0..KSTEP-1, mult of 8*/) {
    int panel  = k >> 5;
    int klocal = k & 31;
    int lhalf  = klocal >> 4;
    int jbase  = klocal & 15;
    return panel * 1024 + (m * 2 + lhalf) * 16 + jbase;
}

__device__ __forceinline__ v8bf pack8(float4 a, float4 b) {
    v8bf r;
    r[0] = (__bf16)a.x; r[1] = (__bf16)a.y; r[2] = (__bf16)a.z; r[3] = (__bf16)a.w;
    r[4] = (__bf16)b.x; r[5] = (__bf16)b.y; r[6] = (__bf16)b.z; r[7] = (__bf16)b.w;
    return r;
}

// Stage one 32m x 128k A panel (f32 global -> bf16 LDS fragment layout).
__device__ __forceinline__ void stage_A(const float* __restrict__ src, int lda,
                                        __bf16* ldsA, int buf, int t) {
    #pragma unroll
    for (int i = 0; i < 2; ++i) {
        int g  = t + i * 256;
        int m  = g >> 4;
        int k8 = (g & 15) * 8;
        const float4* s = (const float4*)&src[(size_t)m * lda + k8];
        *(v8bf*)&ldsA[buf * 4096 + a_lds_group_idx(m, k8)] = pack8(s[0], s[1]);
    }
}

// FWHT(128) per row over ldsC[32][128]: stages 1..8 register-local (8 thr/row,
// 16 contiguous cols each), stages 16/32/64 via LDS; scale folded into h=64.
__device__ __forceinline__ void fwht_epilogue(float* ldsC, int t) {
    {
        int r = t >> 3, c0 = (t & 7) * 16;
        float f[16];
        #pragma unroll
        for (int q = 0; q < 4; ++q)
            *(float4*)&f[q * 4] = *(const float4*)&ldsC[r * 128 + c0 + q * 4];
        #pragma unroll
        for (int h = 1; h < 16; h <<= 1) {
            #pragma unroll
            for (int pr = 0; pr < 8; ++pr) {
                int i = ((pr & ~(h - 1)) << 1) | (pr & (h - 1));
                float a = f[i], b = f[i + h];
                f[i] = a + b; f[i + h] = a - b;
            }
        }
        #pragma unroll
        for (int q = 0; q < 4; ++q)
            *(float4*)&ldsC[r * 128 + c0 + q * 4] = *(const float4*)&f[q * 4];
    }
    __syncthreads();
    #pragma unroll
    for (int h = 16; h < 128; h <<= 1) {
        const float sc = (h == 64) ? FWHT_SCALE : 1.0f;
        #pragma unroll
        for (int p = 0; p < 8; ++p) {
            int g  = t + p * 256;
            int r  = g >> 6;
            int pr = g & 63;
            int i  = ((pr & ~(h - 1)) << 1) | (pr & (h - 1));
            float a = ldsC[r * 128 + i];
            float b = ldsC[r * 128 + i + h];
            ldsC[r * 128 + i]     = (a + b) * sc;
            ldsC[r * 128 + i + h] = (a - b) * sc;
        }
        __syncthreads();
    }
}

// ---------------- K kernel: x @ wk_w + b -> LN -> rope -> FWHT ----------------
__launch_bounds__(256)
__global__ void k_proj_kernel(const float* __restrict__ x,
                              const __bf16* __restrict__ wk_t,
                              const float* __restrict__ wk_b,
                              const float* __restrict__ ln_g,
                              const float* __restrict__ ln_b,
                              const float* __restrict__ freqs,
                              float* __restrict__ outK) {
    __shared__ __align__(32) __bf16 ldsA[2 * 4 * 32 * 2 * 16];   // double-buffered, 16 KB
    __shared__ float ldsC[32 * 128];                              // 16 KB
    __shared__ float red[32 * 8 * 2];

    const int t    = threadIdx.x;
    const int lane = t & 31;
    const int wave = t >> 5;
    const int m0   = blockIdx.x * 32;
    const int ncol = wave * 16 + (lane & 15);
    const int lh16 = (lane >> 4) * 16;
    const float* xrow = x + (size_t)m0 * DIM_;

    v8f acc0 = {}; v8f acc1 = {};

    const int NIT = DIM_ / KSTEP;   // 16
    stage_A(xrow, DIM_, ldsA, 0, t);
    for (int it = 0; it < NIT; ++it) {
        __syncthreads();
        if (it + 1 < NIT) stage_A(xrow + (it + 1) * KSTEP, DIM_, ldsA, (it + 1) & 1, t);
        const int buf = (it & 1) * 4096;
        const int k0  = it * KSTEP;
        #pragma unroll
        for (int p = 0; p < 4; ++p) {
            v16bf bfrag = *(const v16bf*)&wk_t[((size_t)((k0 >> 5) + p) * HD_ + ncol) * 32 + lh16];
            v16bf af0 = *(const v16bf*)&ldsA[buf + p * 1024 + (((lane & 15)) * 2 + (lane >> 4)) * 16];
            v16bf af1 = *(const v16bf*)&ldsA[buf + p * 1024 + ((16 + (lane & 15)) * 2 + (lane >> 4)) * 16];
            acc0 = __builtin_amdgcn_wmma_f32_16x16x32_bf16(false, af0, false, bfrag, (short)0, acc0, false, false);
            acc1 = __builtin_amdgcn_wmma_f32_16x16x32_bf16(false, af1, false, bfrag, (short)0, acc1, false, false);
        }
    }

    __syncthreads();
    {   // C/D layout: VGPR i -> row i + 8*(lane>>4); col = lane&15
        int col   = wave * 16 + (lane & 15);
        int rbase = (lane >> 4) * 8;
        #pragma unroll
        for (int i = 0; i < 8; ++i) {
            ldsC[(rbase + i) * 128 + col]      = acc0[i];
            ldsC[(16 + rbase + i) * 128 + col] = acc1[i];
        }
    }
    __syncthreads();

    // bias + LN partials
    {
        int r = t >> 3, c0 = (t & 7) * 16;
        float s1 = 0.f, s2 = 0.f;
        #pragma unroll
        for (int i = 0; i < 16; ++i) {
            float v = ldsC[r * 128 + c0 + i] + wk_b[c0 + i];
            ldsC[r * 128 + c0 + i] = v;
            s1 += v; s2 += v * v;
        }
        red[(r * 8 + (t & 7)) * 2]     = s1;
        red[(r * 8 + (t & 7)) * 2 + 1] = s2;
    }
    __syncthreads();
    {
        int r = t >> 3, c0 = (t & 7) * 16;
        float s1 = 0.f, s2 = 0.f;
        #pragma unroll
        for (int i = 0; i < 8; ++i) { s1 += red[(r * 8 + i) * 2]; s2 += red[(r * 8 + i) * 2 + 1]; }
        float mu  = s1 * (1.f / 128.f);
        float var = s2 * (1.f / 128.f) - mu * mu;
        float inv = rsqrtf(var + 1e-5f);
        #pragma unroll
        for (int i = 0; i < 16; ++i) {
            int c = c0 + i;
            ldsC[r * 128 + c] = (ldsC[r * 128 + c] - mu) * inv * ln_g[c] + ln_b[c];
        }
    }
    __syncthreads();

    // rope on cols [0,64)
    #pragma unroll
    for (int p = 0; p < 4; ++p) {
        int g = t * 4 + p;
        int r = g >> 5, j = g & 31;
        int s_pos = (m0 + r) & (S_ - 1);
        float cs = freqs[(size_t)(s_pos * 32 + j) * 2];
        float sn = freqs[(size_t)(s_pos * 32 + j) * 2 + 1];
        float x1 = ldsC[r * 128 + j];
        float x2 = ldsC[r * 128 + 32 + j];
        ldsC[r * 128 + j]      = x1 * cs - x2 * sn;
        ldsC[r * 128 + 32 + j] = x1 * sn + x2 * cs;
    }
    __syncthreads();

    fwht_epilogue(ldsC, t);

    // tile is contiguous 16KB in outK: async copy LDS -> global
#if HAVE_ASYNC_ST
    float* gbase = outK + (size_t)m0 * 128;
    #pragma unroll
    for (int i = 0; i < 4; ++i) {
        int e = (t + i * 256) * 4;
        __builtin_amdgcn_global_store_async_from_lds_b128(
            (AS1 v4i*)&gbase[e], (AS3 v4i*)&ldsC[e], 0, 0);
    }
    async_drain();
#else
    #pragma unroll
    for (int i = 0; i < 16; ++i) {
        int e = t + i * 256;
        outK[(size_t)m0 * 128 + e] = ldsC[e];
    }
#endif
}

// ---------------- Q kernel: Q_latent @ wq_w[:,head] + b -> rope -> FWHT -------
__launch_bounds__(256)
__global__ void q_proj_kernel(const float* __restrict__ ql,
                              const __bf16* __restrict__ wq_t,
                              const float* __restrict__ wq_b,
                              const float* __restrict__ freqs,
                              float* __restrict__ outQ) {
    __shared__ __align__(32) __bf16 ldsA[2 * 4 * 32 * 2 * 16];
    __shared__ float ldsC[32 * 128];

    const int t    = threadIdx.x;
    const int lane = t & 31;
    const int wave = t >> 5;
    const int m0   = blockIdx.x * 32;
    const int head = blockIdx.y;
    const int ncol = head * HD_ + wave * 16 + (lane & 15);
    const int lh16 = (lane >> 4) * 16;
    const float* arow = ql + (size_t)m0 * QLR_;

    v8f acc0 = {}; v8f acc1 = {};

    const int NIT = QLR_ / KSTEP;   // 4
    stage_A(arow, QLR_, ldsA, 0, t);
    for (int it = 0; it < NIT; ++it) {
        __syncthreads();
        if (it + 1 < NIT) stage_A(arow + (it + 1) * KSTEP, QLR_, ldsA, (it + 1) & 1, t);
        const int buf = (it & 1) * 4096;
        const int k0  = it * KSTEP;
        #pragma unroll
        for (int p = 0; p < 4; ++p) {
            v16bf bfrag = *(const v16bf*)&wq_t[((size_t)((k0 >> 5) + p) * DIM_ + ncol) * 32 + lh16];
            v16bf af0 = *(const v16bf*)&ldsA[buf + p * 1024 + (((lane & 15)) * 2 + (lane >> 4)) * 16];
            v16bf af1 = *(const v16bf*)&ldsA[buf + p * 1024 + ((16 + (lane & 15)) * 2 + (lane >> 4)) * 16];
            acc0 = __builtin_amdgcn_wmma_f32_16x16x32_bf16(false, af0, false, bfrag, (short)0, acc0, false, false);
            acc1 = __builtin_amdgcn_wmma_f32_16x16x32_bf16(false, af1, false, bfrag, (short)0, acc1, false, false);
        }
    }

    __syncthreads();
    {
        int col   = wave * 16 + (lane & 15);
        int rbase = (lane >> 4) * 8;
        #pragma unroll
        for (int i = 0; i < 8; ++i) {
            ldsC[(rbase + i) * 128 + col]      = acc0[i];
            ldsC[(16 + rbase + i) * 128 + col] = acc1[i];
        }
    }
    __syncthreads();

    // bias
    #pragma unroll
    for (int i = 0; i < 16; ++i) {
        int e = t + i * 256;
        int r = e >> 7, c = e & 127;
        ldsC[r * 128 + c] += wq_b[head * HD_ + c];
    }
    __syncthreads();

    // rope on cols [0,64) of this head
    #pragma unroll
    for (int p = 0; p < 4; ++p) {
        int g = t * 4 + p;
        int r = g >> 5, j = g & 31;
        int s_pos = (m0 + r) & (S_ - 1);
        float cs = freqs[(size_t)(s_pos * 32 + j) * 2];
        float sn = freqs[(size_t)(s_pos * 32 + j) * 2 + 1];
        float x1 = ldsC[r * 128 + j];
        float x2 = ldsC[r * 128 + 32 + j];
        ldsC[r * 128 + j]      = x1 * cs - x2 * sn;
        ldsC[r * 128 + 32 + j] = x1 * sn + x2 * cs;
    }
    __syncthreads();

    fwht_epilogue(ldsC, t);

#if HAVE_ASYNC_ST
    float* gbase = outQ + (size_t)m0 * (H_ * HD_) + head * HD_;
    #pragma unroll
    for (int i = 0; i < 4; ++i) {
        int e = (t + i * 256) * 4;     // float index in tile
        int r = e >> 7, c = e & 127;
        __builtin_amdgcn_global_store_async_from_lds_b128(
            (AS1 v4i*)&gbase[(size_t)r * (H_ * HD_) + c], (AS3 v4i*)&ldsC[e], 0, 0);
    }
    async_drain();
#else
    #pragma unroll
    for (int i = 0; i < 16; ++i) {
        int e = t + i * 256;
        int r = e >> 7, c = e & 127;
        outQ[(size_t)(m0 + r) * (H_ * HD_) + head * HD_ + c] = ldsC[r * 128 + c];
    }
#endif
}

extern "C" void kernel_launch(void* const* d_in, const int* in_sizes, int n_in,
                              void* d_out, int out_size, void* d_ws, size_t ws_size,
                              hipStream_t stream) {
    const float* x        = (const float*)d_in[0];
    const float* Q_latent = (const float*)d_in[1];
    const float* freqs    = (const float*)d_in[2];
    const float* wq_w     = (const float*)d_in[3];
    const float* wq_b     = (const float*)d_in[4];
    const float* wk_w     = (const float*)d_in[5];
    const float* wk_b     = (const float*)d_in[6];
    const float* ln_g     = (const float*)d_in[7];
    const float* ln_b     = (const float*)d_in[8];

    __bf16* wq_t = (__bf16*)d_ws;                        // 2 MB
    __bf16* wk_t = wq_t + (size_t)QLR_ * DIM_;           // +512 KB

    convert_w_kernel<<<(QLR_ * DIM_ + 255) / 256, 256, 0, stream>>>(wq_w, wq_t, QLR_, DIM_);
    convert_w_kernel<<<(DIM_ * HD_ + 255) / 256, 256, 0, stream>>>(wk_w, wk_t, DIM_, HD_);

    float* outQ = (float*)d_out;
    float* outK = outQ + (size_t)ROWS * H_ * HD_;

    dim3 gq(ROWS / 32, H_);
    q_proj_kernel<<<gq, 256, 0, stream>>>(Q_latent, wq_t, wq_b, freqs, outQ);
    k_proj_kernel<<<ROWS / 32, 256, 0, stream>>>(x, wk_t, wk_b, ln_g, ln_b, freqs, outK);
}
